// LongformerAttentionMethod_62199716380751
// MI455X (gfx1250) — matrix operand order repaired and checked by reading, not compiled, not run on hardware
//
#include <hip/hip_runtime.h>
#include <hip/hip_bf16.h>

// ---- problem constants (match reference) ----
#define BH   24        // B*H
#define SEQ  4096
#define DH   64
#define WW   128       // window
#define NB   (SEQ / WW)   // 32 query blocks
#define BAND 257          // 2W+1
#define KROWS 384         // 3W keys per block
#define KSTR  72          // LDS K row stride (halves), padded
#define VSTR  392         // LDS V^T / S row stride (halves), padded

typedef _Float16 v16h __attribute__((ext_vector_type(16)));
typedef _Float16 h4   __attribute__((ext_vector_type(4)));
typedef float    v8f  __attribute__((ext_vector_type(8)));
typedef int      v4i  __attribute__((ext_vector_type(4)));

typedef __attribute__((address_space(1))) v4i* gv4i_p;   // global int4*
typedef __attribute__((address_space(3))) v4i* lv4i_p;   // LDS int4*

#if __has_builtin(__builtin_amdgcn_global_load_async_to_lds_b128) && \
    __has_builtin(__builtin_amdgcn_s_wait_asynccnt)
#define USE_ASYNC_LDS 1
#else
#define USE_ASYNC_LDS 0
#endif

// A-matrix (16x32 f16) element e (0..15) -> K index, given lane half (lane>>4).
// ISA 7.12.2: VGPR v=e/2; v<4: K = half*8 + 2v + e%2 ; v>=4: K = 16 + half*8 + 2(v-4) + e%2
__device__ __forceinline__ int kdOf(int e, int half) {
  int v = e >> 1;
  return ((v & 4) ? 16 : 0) + half * 8 + (v & 3) * 2 + (e & 1);
}

__global__ __launch_bounds__(256)
void longformer_wmma_kernel(const float* __restrict__ q,
                            const float* __restrict__ k,
                            const float* __restrict__ v,
                            float* __restrict__ ctx,
                            float* __restrict__ probs) {
  __shared__ __attribute__((aligned(16))) _Float16 ldsK[KROWS * KSTR]; // K3 [y][d]  (55296 B)
  __shared__ __attribute__((aligned(16))) _Float16 ldsVt[DH * VSTR];   // V3^T [d][y](50176 B)
  __shared__ __attribute__((aligned(16))) _Float16 ldsS[64 * VSTR];    // S/P strip  (50176 B)

  const int n   = blockIdx.x;     // query block 0..31
  const int bh  = blockIdx.y;     // 0..23
  const int tid = threadIdx.x;    // 0..255 (8 wave32)
  const int w      = tid >> 5;
  const int lane   = tid & 31;
  const int lane16 = lane & 15;
  const int half   = lane >> 4;
  const int jbase  = n * WW - WW; // global key index of y=0

  // token prefetch of next query block (global_prefetch_b8 path)
  if (tid == 0 && n + 1 < NB) {
    __builtin_prefetch(q + ((size_t)bh * SEQ + (n + 1) * WW) * DH, 0, 1);
  }

  // ---------------- Stage A: K3 / V3 -> LDS (f32 -> f16, zero pad) --------
#if USE_ASYNC_LDS
  {
    float* stage = (float*)ldsS;   // 192 rows x 64 f32 = 49152 B staging window
    for (int chunk = 0; chunk < 4; ++chunk) {
      const int mbase = (chunk & 1) * 192;
      const float* src = (chunk < 2) ? k : v;
      // issue ASYNCcnt-tracked DMA: global f32 -> LDS staging
      for (int gg = tid; gg < 3072; gg += 256) {
        int y = gg >> 4, d0 = (gg & 15) << 2;
        int j = jbase + mbase + y;
        if (j >= 0 && j < SEQ) {
          __builtin_amdgcn_global_load_async_to_lds_b128(
              (gv4i_p)(src + ((size_t)bh * SEQ + j) * DH + d0),
              (lv4i_p)(stage + y * DH + d0),
              0, 0);
        }
      }
      __builtin_amdgcn_s_wait_asynccnt(0);
      __syncthreads();
      // convert staging f32 -> destination f16 (zeros for padded rows)
      for (int gg = tid; gg < 3072; gg += 256) {
        int y = gg >> 4, d0 = (gg & 15) << 2;
        int j = jbase + mbase + y;
        float4 t4 = make_float4(0.f, 0.f, 0.f, 0.f);
        if (j >= 0 && j < SEQ) t4 = *(const float4*)(stage + y * DH + d0);
        if (chunk < 2) {
          h4 hh = { (_Float16)t4.x, (_Float16)t4.y, (_Float16)t4.z, (_Float16)t4.w };
          *(h4*)(&ldsK[(mbase + y) * KSTR + d0]) = hh;
        } else {
          ldsVt[(d0 + 0) * VSTR + mbase + y] = (_Float16)t4.x;
          ldsVt[(d0 + 1) * VSTR + mbase + y] = (_Float16)t4.y;
          ldsVt[(d0 + 2) * VSTR + mbase + y] = (_Float16)t4.z;
          ldsVt[(d0 + 3) * VSTR + mbase + y] = (_Float16)t4.w;
        }
      }
      __syncthreads();   // staging reused next chunk
    }
  }
#else
  for (int g = tid; g < (KROWS * DH) / 4; g += 256) {
    int y  = g >> 4;
    int d0 = (g & 15) << 2;
    int j  = jbase + y;
    float4 kv = make_float4(0.f, 0.f, 0.f, 0.f);
    float4 vv = make_float4(0.f, 0.f, 0.f, 0.f);
    if (j >= 0 && j < SEQ) {
      size_t base = ((size_t)bh * SEQ + j) * DH + d0;
      kv = *(const float4*)(k + base);
      vv = *(const float4*)(v + base);
    }
    h4 kh = { (_Float16)kv.x, (_Float16)kv.y, (_Float16)kv.z, (_Float16)kv.w };
    *(h4*)(&ldsK[y * KSTR + d0]) = kh;
    ldsVt[(d0 + 0) * VSTR + y] = (_Float16)vv.x;
    ldsVt[(d0 + 1) * VSTR + y] = (_Float16)vv.y;
    ldsVt[(d0 + 2) * VSTR + y] = (_Float16)vv.z;
    ldsVt[(d0 + 3) * VSTR + y] = (_Float16)vv.w;
  }
  __syncthreads();
#endif

  // ---------------- two phases of 64 query rows ---------------------------
  for (int ph = 0; ph < 2; ++ph) {
    const int r0 = ph * 64;

    // ---- QK^T: wave w owns row-panel (w>>1), column tiles [(w&1)*12, +12)
    {
      const int rp      = w >> 1;
      const int colbase = (w & 1) * 12;
      const int xmin    = r0 + rp * 16;      // first query row of panel
      const int t0      = xmin >> 4;         // band tiles: [t0, t0+16]
      // scalarize the (wave-uniform) band-intersection bounds so the tile
      // loop is an SALU-bounded loop and EXEC stays all-ones around WMMA
      const int ctLo = __builtin_amdgcn_readfirstlane(max(colbase, t0));
      const int ctHi = __builtin_amdgcn_readfirstlane(min(colbase + 11, t0 + 16));
      const int qrow    = n * WW + xmin + lane16;
      const float* qp   = q + ((size_t)bh * SEQ + qrow) * DH;

      v16h a0, a1;
      #pragma unroll
      for (int e = 0; e < 16; ++e) {
        int kd = kdOf(e, half);
        a0[e] = (_Float16)qp[kd];
        a1[e] = (_Float16)qp[32 + kd];
      }

      for (int ct = ctLo; ct <= ctHi; ++ct) {
        const _Float16* kp = &ldsK[(ct * 16 + lane16) * KSTR];
        v16h b0, b1;
        #pragma unroll
        for (int e = 0; e < 16; ++e) {           // B: K = kdOf(e, half) (same as A)
          b0[e] = kp[kdOf(e, half)];
          b1[e] = kp[32 + kdOf(e, half)];
        }
        v8f acc = {0.f, 0.f, 0.f, 0.f, 0.f, 0.f, 0.f, 0.f};
        acc = __builtin_amdgcn_wmma_f32_16x16x32_f16(false, a0, false, b0,
                                                     (short)0, acc, false, false);
        acc = __builtin_amdgcn_wmma_f32_16x16x32_f16(false, a1, false, b1,
                                                     (short)0, acc, false, false);
        #pragma unroll
        for (int jj = 0; jj < 8; ++jj) {
          int M = jj + half * 8;
          ldsS[(rp * 16 + M) * VSTR + ct * 16 + lane16] =
              (_Float16)(acc[jj] * 0.125f);      // * 1/sqrt(64)
        }
      }
    }
    __syncthreads();

    // ---- banded softmax: 4 lanes per row, interleaved slots ---------------
    {
      const int rloc = tid >> 2;          // 0..63
      const int quad = tid & 3;
      const int x    = r0 + rloc;         // in-block query index 0..127
      const int i    = n * WW + x;        // global query index
      _Float16* srow = &ldsS[rloc * VSTR];

      float m = -3.0e38f;
      #pragma unroll 4
      for (int it = 0; it < 96; ++it) {
        int y = quad + 4 * it, j = jbase + y;
        bool ok = (y >= x) && (y <= x + 256) && (j >= 0) && (j < SEQ);
        float s = (float)srow[y];
        m = ok ? fmaxf(m, s) : m;
      }
      m = fmaxf(m, __shfl_xor(m, 1, 32));
      m = fmaxf(m, __shfl_xor(m, 2, 32));

      float sum = 0.f;
      #pragma unroll 4
      for (int it = 0; it < 96; ++it) {
        int y = quad + 4 * it, j = jbase + y;
        bool ok = (y >= x) && (y <= x + 256) && (j >= 0) && (j < SEQ);
        float e = ok ? __expf((float)srow[y] - m) : 0.f;
        srow[y] = (_Float16)e;
        sum += e;
      }
      sum += __shfl_xor(sum, 1, 32);
      sum += __shfl_xor(sum, 2, 32);
      const float inv = 1.0f / sum;

      float* prow = probs + ((size_t)bh * SEQ + i) * BAND;
      #pragma unroll 4
      for (int it = 0; it < 96; ++it) {
        int y = quad + 4 * it;
        float p = (float)srow[y] * inv;
        srow[y] = (_Float16)p;
        int t = y - x;                      // 4 lanes of a row -> 4 consecutive t
        if (t >= 0 && t <= 256) prow[t] = p;
      }
    }
    __syncthreads();

    // ---- PV: 16 output tiles (4x4); wave w does tiles 2w, 2w+1 ------------
    for (int u = 0; u < 2; ++u) {
      const int ti    = w * 2 + u;
      const int rp2   = ti >> 2;
      const int cn    = ti & 3;
      const int xmin2 = r0 + rp2 * 16;
      // P is nonzero only for band chunks: scalar contiguous kk range
      const int kkLo  = __builtin_amdgcn_readfirstlane(xmin2 >> 5);
      const int kkHi  = __builtin_amdgcn_readfirstlane((xmin2 + 271) >> 5);
      const _Float16* pr  = &ldsS[(rp2 * 16 + lane16) * VSTR];
      const _Float16* vtp = &ldsVt[(cn * 16 + lane16) * VSTR];
      v8f acc = {0.f, 0.f, 0.f, 0.f, 0.f, 0.f, 0.f, 0.f};
      for (int kk = kkLo; kk <= kkHi; ++kk) {
        v16h pa, vb;
        #pragma unroll
        for (int e = 0; e < 16; ++e) {
          pa[e] = pr[kk * 32 + kdOf(e, half)];
          vb[e] = vtp[kk * 32 + kdOf(e, half)];
        }
        acc = __builtin_amdgcn_wmma_f32_16x16x32_f16(false, pa, false, vb,
                                                     (short)0, acc, false, false);
      }
      #pragma unroll
      for (int jj = 0; jj < 8; ++jj) {
        int M = jj + half * 8;
        int i = n * WW + r0 + rp2 * 16 + M;
        ctx[((size_t)bh * SEQ + i) * DH + cn * 16 + lane16] = acc[jj];
      }
    }
    __syncthreads();   // before next phase reuses ldsS
  }
}

extern "C" void kernel_launch(void* const* d_in, const int* in_sizes, int n_in,
                              void* d_out, int out_size, void* d_ws, size_t ws_size,
                              hipStream_t stream) {
  (void)in_sizes; (void)n_in; (void)d_ws; (void)ws_size; (void)out_size;
  const float* q = (const float*)d_in[0];
  const float* k = (const float*)d_in[1];
  const float* v = (const float*)d_in[2];
  float* ctx   = (float*)d_out;
  float* probs = (float*)d_out + (size_t)BH * SEQ * DH;   // context first, probs second
  dim3 grid(NB, BH);
  longformer_wmma_kernel<<<grid, 256, 0, stream>>>(q, k, v, ctx, probs);
}
